// DictionaryLearning_21019569946795
// MI455X (gfx1250) — matrix operand order repaired and checked by reading, not compiled, not run on hardware
//
#include <hip/hip_runtime.h>
#include <hip/hip_bf16.h>
#include <stdint.h>

// ---------------------------------------------------------------------------
// Problem constants (from reference)
// ---------------------------------------------------------------------------
#define BATCH   8
#define CHN     64
#define HWDIM   128
#define HO      63
#define WO      63
#define NPATCH  (BATCH * HO * WO)     // 31752
#define PDIM    1024                  // 64 * 4 * 4
#define NATOMS  512
#define SPARS   5
#define EPSV    1e-10f

#define WAVES_PER_BLOCK 8
#define MBLOCKS 249                   // ceil(ceil(31752/16)/8)
#define NPAD    (MBLOCKS * WAVES_PER_BLOCK * 16)  // 31872 padded patch rows

#define NGROUP  64                    // atoms per block (4 x 16-wide WMMA tiles)
#define NTILES_PER_WAVE 4

typedef __attribute__((ext_vector_type(16))) __bf16          v16bf;
typedef __attribute__((ext_vector_type(16))) unsigned short  v16us;
typedef __attribute__((ext_vector_type(8)))  float           v8f;

// Split an fp32 value into bf16 hi + bf16 lo (round-to-nearest-even).
// hi + lo reproduces f to ~2^-17 relative accuracy.
__device__ __forceinline__ void split_bf16(float f, unsigned short& hi, unsigned short& lo) {
    unsigned int u = __float_as_uint(f);
    unsigned int r = u + 0x7FFFu + ((u >> 16) & 1u);
    hi = (unsigned short)(r >> 16);
    float fh = __uint_as_float(((unsigned int)hi) << 16);
    float fl = f - fh;
    unsigned int u2 = __float_as_uint(fl);
    unsigned int r2 = u2 + 0x7FFFu + ((u2 >> 16) & 1u);
    lo = (unsigned short)(r2 >> 16);
}

// ---------------------------------------------------------------------------
// Kernel: gram = D^T D  (512x512, K=1024). 0.54 GFLOP -> plain FMA is fine.
// i is block-uniform (scalar load), j coalesced.
// ---------------------------------------------------------------------------
__global__ __launch_bounds__(256) void gram_kernel(const float* __restrict__ dict,
                                                   float* __restrict__ gram) {
    int e = blockIdx.x * 256 + threadIdx.x;
    int i = e >> 9;
    int j = e & 511;
    float s = 0.f;
    for (int k = 0; k < PDIM; ++k)
        s = fmaf(dict[(size_t)k * NATOMS + i], dict[(size_t)k * NATOMS + j], s);
    gram[e] = s;
}

// ---------------------------------------------------------------------------
// Kernel: corr = X @ D via v_wmma_f32_16x16x32_bf16 with hi/lo split (3 WMMA
// per K-step per N-tile). Fused im2col: A rows are gathered straight from x,
// split once per K-step, then reused across 4 N-tiles (12 WMMAs) so the
// gather+split VALU cost is amortized against the matrix pipe.
// Block: 8 waves = 8 M-tiles (128 patches) x one 64-atom N-group.
// ---------------------------------------------------------------------------
__global__ __launch_bounds__(256) void corr_wmma_kernel(const float* __restrict__ x,
                                                        const float* __restrict__ dict,
                                                        float* __restrict__ corr) {
    __shared__ unsigned short sBh[NGROUP * 32];   // [n][k] layout, bf16-hi (4 KB)
    __shared__ unsigned short sBl[NGROUP * 32];   // [n][k] layout, bf16-lo (4 KB)

    const int tid    = threadIdx.x;
    const int lane   = tid & 31;
    const int wave   = tid >> 5;
    const int n0     = blockIdx.x * NGROUP;                   // atom group base
    const int m_tile = blockIdx.y * WAVES_PER_BLOCK + wave;   // 16-patch tile

    // ---- per-lane A-row geometry (fused im2col) ----
    const int  l     = lane & 15;
    const int  gm    = m_tile * 16 + l;                       // global patch row
    const bool valid = (gm < NPATCH);
    int b = 0, ho = 0, wo = 0;
    if (valid) { b = gm / (HO * WO); int rem = gm % (HO * WO); ho = rem / WO; wo = rem % WO; }
    const long xbase  = (long)b * (CHN * HWDIM * HWDIM) + (long)(ho * 2) * HWDIM + (wo * 2);
    const int  khalfA = (lane >> 4) ? 8 : 0;   // ISA A layout: hi-half lanes hold K+8

    // ---- B fragment geometry ----
    const int nB  = lane & 15;                 // N within a 16-wide tile
    const int khB = (lane >> 4) ? 8 : 0;       // hi-half lanes hold K+8 (same layout as A)

    v8f acc[NTILES_PER_WAVE] = {};

    for (int k0 = 0; k0 < PDIM; k0 += 32) {
        // Stage + split D slice (32 K x 64 N) into LDS, transposed to [n][k].
        // Consecutive threads touch consecutive atoms -> coalesced global reads.
        #pragma unroll
        for (int i = 0; i < (32 * NGROUP) / 256; ++i) {
            const int e  = i * 256 + tid;
            const int kk = e >> 6;          // 0..31
            const int nn = e & 63;          // 0..63
            float g = dict[(size_t)(k0 + kk) * NATOMS + n0 + nn];
            unsigned short hi, lo;
            split_bf16(g, hi, lo);
            sBh[nn * 32 + kk] = hi;
            sBl[nn * 32 + kk] = lo;
        }
        __syncthreads();
        if (k0 + 32 < PDIM)
            __builtin_prefetch(&dict[(size_t)(k0 + 32) * NATOMS + n0], 0, 0);

        // Build A fragments once per K-step (gather from x, split hi/lo).
        // 16-bit A 16x32 layout: lanes 0-15: VGPR j<4 -> K=2j,2j+1 ; j>=4 -> K=16+2(j-4)..
        // lanes 16-31: same +8.
        v16us ah = {}, al = {};
        #pragma unroll
        for (int j = 0; j < 8; ++j) {
            const int krel = ((j < 4) ? 0 : 16) + khalfA + 2 * (j & 3);
            #pragma unroll
            for (int t = 0; t < 2; ++t) {
                const int kabs = k0 + krel + t;
                float f = 0.f;
                if (valid) {
                    int c  = kabs >> 4;          // channel
                    int r  = (kabs >> 2) & 3;    // patch row
                    int cc = kabs & 3;           // patch col
                    f = x[xbase + (long)c * (HWDIM * HWDIM) + r * HWDIM + cc];
                }
                unsigned short hi, lo;
                split_bf16(f, hi, lo);
                ah[2 * j + t] = hi;
                al[2 * j + t] = lo;
            }
        }
        const v16bf ahb = __builtin_bit_cast(v16bf, ah);
        const v16bf alb = __builtin_bit_cast(v16bf, al);

        // Reuse A across 4 N-tiles: 12 WMMAs per K-step.
        #pragma unroll
        for (int nt = 0; nt < NTILES_PER_WAVE; ++nt) {
            const int nbase = (nt * 16 + nB) * 32;
            v16us bh = {}, bl = {};
            #pragma unroll
            for (int j = 0; j < 8; ++j) {
                const int K = ((j < 4) ? 0 : 16) + khB + 2 * (j & 3);
                bh[2 * j]     = sBh[nbase + K];
                bh[2 * j + 1] = sBh[nbase + K + 1];
                bl[2 * j]     = sBl[nbase + K];
                bl[2 * j + 1] = sBl[nbase + K + 1];
            }
            const v16bf bhb = __builtin_bit_cast(v16bf, bh);
            const v16bf blb = __builtin_bit_cast(v16bf, bl);

            // hi*hi + hi*lo + lo*hi  ->  ~fp32-accurate product at WMMA rate.
            acc[nt] = __builtin_amdgcn_wmma_f32_16x16x32_bf16(
                false, ahb, false, bhb, (short)0, acc[nt], false, false);
            acc[nt] = __builtin_amdgcn_wmma_f32_16x16x32_bf16(
                false, ahb, false, blb, (short)0, acc[nt], false, false);
            acc[nt] = __builtin_amdgcn_wmma_f32_16x16x32_bf16(
                false, alb, false, bhb, (short)0, acc[nt], false, false);
        }

        __syncthreads();  // protect LDS before next stage
    }

    // C layout: VGPR v -> (M = v, lanes 0-15) / (M = v+8, lanes 16-31), N = lane%16.
    const int    col    = lane & 15;
    const int    rowoff = (lane >> 4) ? 8 : 0;
    const size_t mbase  = (size_t)m_tile * 16;
    #pragma unroll
    for (int nt = 0; nt < NTILES_PER_WAVE; ++nt) {
        #pragma unroll
        for (int v = 0; v < 8; ++v)
            corr[(mbase + v + rowoff) * NATOMS + n0 + nt * 16 + col] = acc[nt][v];
    }
}

// ---------------------------------------------------------------------------
// Kernel: per-patch top-5 of |corr| (wave32 shuffle argmax, tie-break = lowest
// index to match lax.top_k) + 5x5 SPD solve (Gaussian elimination, lane 0).
// One wave per patch, 8 waves per block.
// ---------------------------------------------------------------------------
__global__ __launch_bounds__(256) void topk_solve_kernel(const float* __restrict__ corr,
                                                         const float* __restrict__ gram,
                                                         float* __restrict__ vals,
                                                         int* __restrict__ sup) {
    const int lane = threadIdx.x & 31;
    const int p    = blockIdx.x * 8 + (threadIdx.x >> 5);
    if (p >= NPATCH) return;   // no barriers in this kernel -> safe early exit

    const float* crow = corr + (size_t)p * NATOMS;
    float v[16], a[16];
    #pragma unroll
    for (int i = 0; i < 16; ++i) {
        v[i] = crow[i * 32 + lane];
        a[i] = fabsf(v[i]);
    }

    int   selI[SPARS];
    float selV[SPARS];
    for (int s = 0; s < SPARS; ++s) {
        float bm = -1.f, bv = 0.f;
        int   bi = 0x7fffffff;
        #pragma unroll
        for (int i = 0; i < 16; ++i) {
            const int ix = i * 32 + lane;
            if (a[i] > bm || (a[i] == bm && ix < bi)) { bm = a[i]; bv = v[i]; bi = ix; }
        }
        #pragma unroll
        for (int off = 16; off >= 1; off >>= 1) {
            float om = __shfl_xor(bm, off);
            float ov = __shfl_xor(bv, off);
            int   oi = __shfl_xor(bi, off);
            if (om > bm || (om == bm && oi < bi)) { bm = om; bv = ov; bi = oi; }
        }
        selI[s] = bi;
        selV[s] = bv;
        // mask the winner (static indexing only)
        #pragma unroll
        for (int i = 0; i < 16; ++i)
            if (bi == i * 32 + lane) a[i] = -2.f;
    }

    if (lane == 0) {
        float G[SPARS][SPARS], rv[SPARS];
        for (int i = 0; i < SPARS; ++i) {
            rv[i] = selV[i];
            for (int j = 0; j < SPARS; ++j)
                G[i][j] = gram[(size_t)selI[i] * NATOMS + selI[j]] + (i == j ? EPSV : 0.f);
        }
        // SPD (gram + eps*I): elimination without pivoting.
        for (int k = 0; k < SPARS; ++k) {
            const float inv = 1.f / G[k][k];
            for (int j = k; j < SPARS; ++j) G[k][j] *= inv;
            rv[k] *= inv;
            for (int i = k + 1; i < SPARS; ++i) {
                const float f = G[i][k];
                for (int j = k; j < SPARS; ++j) G[i][j] -= f * G[k][j];
                rv[i] -= f * rv[k];
            }
        }
        for (int i = SPARS - 2; i >= 0; --i)
            for (int j = i + 1; j < SPARS; ++j) rv[i] -= G[i][j] * rv[j];

        for (int s = 0; s < SPARS; ++s) {
            vals[p * SPARS + s] = rv[s];
            sup [p * SPARS + s] = selI[s];
        }
    }
}

// ---------------------------------------------------------------------------
// Kernel: reconstruction + loss. hann(4) = [0,.75,.75,0] => each interior
// pixel is covered by exactly one (patch, r, cc) with r,cc in {1,2} and
// num/den == X_hat exactly; border pixels (h or w in {0,127}) get 0.
// quantized == z_q numerically. Accumulate sum((z_q - x)^2).
// ---------------------------------------------------------------------------
__global__ __launch_bounds__(256) void recon_kernel(const float* __restrict__ x,
                                                    const float* __restrict__ dict,
                                                    const float* __restrict__ vals,
                                                    const int* __restrict__ sup,
                                                    float* __restrict__ out,
                                                    float* __restrict__ acc) {
    __shared__ float red[256];
    const int idx = blockIdx.x * 256 + threadIdx.x;   // exact multiple of 256
    const int w = idx & 127;
    const int h = (idx >> 7) & 127;
    const int c = (idx >> 14) & 63;
    const int b = idx >> 20;

    int ho, r, wo, cc;
    if (h & 1) { ho = (h - 1) >> 1; r = 1; } else { ho = (h >> 1) - 1; r = 2; }
    if (w & 1) { wo = (w - 1) >> 1; cc = 1; } else { wo = (w >> 1) - 1; cc = 2; }

    float zq = 0.f;
    if (ho >= 0 && ho < HO && wo >= 0 && wo < WO) {
        const int p   = (b * HO + ho) * WO + wo;
        const int row = c * 16 + r * 4 + cc;
        const float* drow = dict + (size_t)row * NATOMS;
        const float* vp   = vals + (size_t)p * SPARS;
        const int*   sp   = sup  + (size_t)p * SPARS;
        #pragma unroll
        for (int s = 0; s < SPARS; ++s)
            zq = fmaf(vp[s], drow[sp[s]], zq);
    }
    out[idx] = zq;                         // quantized = x + (z_q - x) = z_q
    const float d = zq - x[idx];
    red[threadIdx.x] = d * d;
    __syncthreads();
    for (int o = 128; o > 0; o >>= 1) {
        if (threadIdx.x < o) red[threadIdx.x] += red[threadIdx.x + o];
        __syncthreads();
    }
    if (threadIdx.x == 0) atomicAdd(acc, red[0]);
}

__global__ void init_kernel(float* acc) { acc[0] = 0.f; }

__global__ void finalize_kernel(const float* __restrict__ acc, float* __restrict__ out_loss) {
    // loss = dl_latent + 0.25 * e_latent = 1.25 * mean((z_q - x)^2)
    out_loss[0] = 1.25f * acc[0] / (float)(BATCH * CHN * HWDIM * HWDIM);
}

// ---------------------------------------------------------------------------
// Launcher
// ---------------------------------------------------------------------------
extern "C" void kernel_launch(void* const* d_in, const int* in_sizes, int n_in,
                              void* d_out, int out_size, void* d_ws, size_t ws_size,
                              hipStream_t stream) {
    (void)in_sizes; (void)n_in; (void)out_size; (void)ws_size;
    const float* x    = (const float*)d_in[0];   // (8,64,128,128) fp32
    const float* dict = (const float*)d_in[1];   // (1024,512) fp32
    float*       out  = (float*)d_out;           // quantized (8,64,128,128) ++ loss

    // Workspace layout (~68 MB)
    float* corr = (float*)d_ws;                       // NPAD * 512 fp32 (padded rows)
    float* gram = corr + (size_t)NPAD * NATOMS;       // 512 * 512 fp32
    float* vals = gram + (size_t)NATOMS * NATOMS;     // NPATCH * 5 fp32
    int*   sup  = (int*)(vals + (size_t)NPATCH * SPARS);
    float* acc  = (float*)(sup + (size_t)NPATCH * SPARS);

    hipLaunchKernelGGL(init_kernel, dim3(1), dim3(1), 0, stream, acc);
    hipLaunchKernelGGL(gram_kernel, dim3((NATOMS * NATOMS) / 256), dim3(256), 0, stream,
                       dict, gram);
    hipLaunchKernelGGL(corr_wmma_kernel, dim3(NATOMS / NGROUP, MBLOCKS), dim3(256), 0, stream,
                       x, dict, corr);
    hipLaunchKernelGGL(topk_solve_kernel, dim3((NPATCH + 7) / 8), dim3(256), 0, stream,
                       corr, gram, vals, sup);
    const int total = BATCH * CHN * HWDIM * HWDIM;
    hipLaunchKernelGGL(recon_kernel, dim3(total / 256), dim3(256), 0, stream,
                       x, dict, vals, sup, out, acc);
    hipLaunchKernelGGL(finalize_kernel, dim3(1), dim3(1), 0, stream, acc, out + total);
}
